// cheb_interpolation_conv_33526514712805
// MI455X (gfx1250) — compile-verified, matrix operand order checked by reading, not run on hardware
//
#include <hip/hip_runtime.h>
#include <hip/hip_bf16.h>

// ---- CDNA5 WMMA vector types (wave32) ----
typedef __attribute__((ext_vector_type(16))) __bf16 v16bf;
typedef __attribute__((ext_vector_type(8)))  __bf16 v8bf;
typedef __attribute__((ext_vector_type(8)))  float  v8f;

#define B_      2
#define N_      4096
#define F_IN_   16
#define F_OUT_  16
#define T_      12
#define K_      3
#define C_PER_B (F_OUT_ * T_)   // 192 columns per batch
#define C_TOT   (B_ * C_PER_B)  // 384 total GEMM columns

// ---------------------------------------------------------------------------
// Kernel 1: lhs = einsum('jio,jk->kio', Gamma, xj)   (tiny, per-block in LDS)
//           zt[k][c][m] = bf16( sum_f x[b,m,f,t] * lhs[k,f,o] )
//           with c = b*192 + o*12 + t  (transposed layout -> WMMA B frags
//           become contiguous 16B loads along m)
// ---------------------------------------------------------------------------
__global__ void __launch_bounds__(256)
cheb_prep_zt(const float* __restrict__ x,
             const float* __restrict__ xj,
             const float* __restrict__ Gamma,
             __hip_bfloat16* __restrict__ zt_raw)
{
    __bf16* zt = reinterpret_cast<__bf16*>(zt_raw);
    __shared__ float lhs_s[K_][F_IN_ * F_OUT_];

    const int tid = threadIdx.x;                 // 256 threads = 16*16 (f,o)
    {
        const float g0 = Gamma[0 * 256 + tid];
        const float g1 = Gamma[1 * 256 + tid];
        const float g2 = Gamma[2 * 256 + tid];
#pragma unroll
        for (int k = 0; k < K_; ++k)
            lhs_s[k][tid] = g0 * xj[0 * K_ + k] + g1 * xj[1 * K_ + k] + g2 * xj[2 * K_ + k];
    }
    __syncthreads();

    const int kb = blockIdx.y;                   // 0..5 -> (k, b)
    const int k  = kb / B_;
    const int b  = kb % B_;
    const int m  = blockIdx.x * 256 + tid;       // 0..4095

    const float* xrow = x + (size_t)(b * N_ + m) * (F_IN_ * T_); // 192 floats

#pragma unroll 1
    for (int t = 0; t < T_; ++t) {
        float xf[F_IN_];
#pragma unroll
        for (int f = 0; f < F_IN_; ++f) xf[f] = xrow[f * T_ + t];
#pragma unroll
        for (int o = 0; o < F_OUT_; ++o) {
            float acc = 0.0f;
#pragma unroll
            for (int f = 0; f < F_IN_; ++f)
                acc += xf[f] * lhs_s[k][f * F_OUT_ + o];
            const int c = b * C_PER_B + o * T_ + t;
            zt[(size_t)(k * C_TOT + c) * N_ + m] = (__bf16)acc;   // coalesced over m
        }
    }
}

// ---------------------------------------------------------------------------
// Kernel 2: out[b,n,o,t] = relu(0.5 * sum_k sum_m cheb[k][m][n] * zt[k][c][m])
// Block: 256 threads (8 waves), owns 32 n-rows x all 384 columns.
//   wave w: rowSub = w&1 (16-row subtile), colGroup = w>>1 (6 col-tiles of 16)
// Per 32-step in m: stage cheb tile (fp32->bf16) transposed into LDS so the
// WMMA A fragment (M=n rows, K=m) is two 16B ds reads per lane; B fragments
// read straight from L2-resident zt (two 16B global reads per lane).
// ---------------------------------------------------------------------------
__global__ void __launch_bounds__(256)
cheb_wmma_gemm(const float* __restrict__ cheb,
               const __hip_bfloat16* __restrict__ zt_raw,
               float* __restrict__ out)
{
    const __bf16* zt = reinterpret_cast<const __bf16*>(zt_raw);

    // [n_local][m_local], pitch 40 (80B, 16B-aligned rows, de-conflicts banks)
    __shared__ __attribute__((aligned(16))) __bf16 Abuf[32][40];

    const int tid      = threadIdx.x;
    const int lane     = tid & 31;
    const int w        = tid >> 5;     // wave 0..7
    const int rowSub   = w & 1;        // 16-row subtile
    const int colGroup = w >> 1;       // 0..3 -> 6 column tiles each
    const int n0       = blockIdx.x * 32;
    const int lrow     = lane & 15;
    const int half     = lane >> 4;    // lane-half K-octet selector

    v8f acc[6] = {};                   // six 16x16 f32 accumulators

    const int mmL = tid >> 3;          // 0..31 : m row of staged tile
    const int nnL = (tid & 7) * 4;     // 0..28 : 4 consecutive n per thread

    for (int k = 0; k < K_; ++k) {
        const float*  Ak = cheb + (size_t)k * N_ * N_;
        const __bf16* Zk = zt   + (size_t)k * C_TOT * N_;

        for (int m0 = 0; m0 < N_; m0 += 32) {
            // coalesced fp32 read of cheb[k][m0+mm][n0+nn .. +3]
            const float4 v = *(const float4*)(Ak + (size_t)(m0 + mmL) * N_ + (n0 + nnL));
            if (m0 + 32 < N_)   // pull next tile toward the caches
                __builtin_prefetch(Ak + (size_t)(m0 + 32 + mmL) * N_ + (n0 + nnL), 0, 3);

            __syncthreads();   // previous iteration's WMMA reads done
            Abuf[nnL + 0][mmL] = (__bf16)v.x;   // transpose: [n][m]
            Abuf[nnL + 1][mmL] = (__bf16)v.y;
            Abuf[nnL + 2][mmL] = (__bf16)v.z;
            Abuf[nnL + 3][mmL] = (__bf16)v.w;
            __syncthreads();

            // A fragment: row M = n, K-octets {half*8, 16+half*8} per ISA layout
            const v8bf* arow = (const v8bf*)(&Abuf[rowSub * 16 + lrow][0]);
            v16bf a;
            ((v8bf*)&a)[0] = arow[half];
            ((v8bf*)&a)[1] = arow[2 + half];

#pragma unroll
            for (int j = 0; j < 6; ++j) {
                const int c = (colGroup * 6 + j) * 16 + lrow;     // B column N = c
                const v8bf* brow = (const v8bf*)(Zk + (size_t)c * N_ + m0);
                v16bf bm;
                ((v8bf*)&bm)[0] = brow[half];
                ((v8bf*)&bm)[1] = brow[2 + half];

                acc[j] = __builtin_amdgcn_wmma_f32_16x16x32_bf16(
                    /*neg_a=*/false, a, /*neg_b=*/false, bm,
                    /*c_mod=*/(short)0, acc[j],
                    /*reuse_a=*/false, /*reuse_b=*/false);
            }
            __syncthreads();
        }
    }

    // Epilogue: C/D layout = elem v -> row (half*8 + v), col = lrow
#pragma unroll
    for (int j = 0; j < 6; ++j) {
        const int ct = colGroup * 6 + j;            // 0..23 column tile
        const int b  = ct / 12;                     // 12 tiles per batch
        const int cb = (ct % 12) * 16 + lrow;       // (o,t) column in [0,192)
        const int nb = n0 + rowSub * 16 + half * 8;
#pragma unroll
        for (int v = 0; v < 8; ++v) {
            const int n = nb + v;
            out[(size_t)(b * N_ + n) * C_PER_B + cb] = fmaxf(0.5f * acc[j][v], 0.0f);
        }
    }
}

// ---------------------------------------------------------------------------
extern "C" void kernel_launch(void* const* d_in, const int* in_sizes, int n_in,
                              void* d_out, int out_size, void* d_ws, size_t ws_size,
                              hipStream_t stream)
{
    const float* x     = (const float*)d_in[0];   // (B, N, F_in, T)
    const float* cheb  = (const float*)d_in[1];   // (K, N, N)
    const float* xj    = (const float*)d_in[2];   // (K, K)
    const float* Gamma = (const float*)d_in[3];   // (K, F_in, F_out)
    float* out = (float*)d_out;                   // (B, N, F_out, T)

    __hip_bfloat16* zt = (__hip_bfloat16*)d_ws;   // [K][C_TOT][N] bf16 = 9.4 MB

    cheb_prep_zt<<<dim3(N_ / 256, K_ * B_), 256, 0, stream>>>(x, xj, Gamma, zt);
    cheb_wmma_gemm<<<dim3(N_ / 32), 256, 0, stream>>>(cheb, zt, out);
    (void)in_sizes; (void)n_in; (void)out_size; (void)ws_size;
}
